// onet_81046032876060
// MI455X (gfx1250) — 8-bit matrix operand order repaired and checked by reading, compiled, not run on hardware
//
#include <hip/hip_runtime.h>
#include <stdint.h>

typedef __attribute__((ext_vector_type(8))) int v8i;

__device__ __forceinline__ int kbaseA(int v, int hi) {
  return 16 * (v >> 1) + 8 * hi + 4 * (v & 1);
}
__device__ __forceinline__ int kbaseB(int v, int hi) {
  return 16 * (v >> 1) + 8 * hi + 4 * (v & 1);
}
__device__ __forceinline__ signed char sgn8(float x) { return x > 0.f ? 1 : -1; }

__device__ __forceinline__ void async_copy16(const void* g, void* l) {
#if defined(__HIP_DEVICE_COMPILE__) && __has_builtin(__builtin_amdgcn_global_load_async_to_lds_b128)
  typedef int vec4i __attribute__((vector_size(4 * sizeof(int))));
  typedef __attribute__((address_space(1))) vec4i as1_vec4i;
  typedef __attribute__((address_space(3))) vec4i as3_vec4i;
  __builtin_amdgcn_global_load_async_to_lds_b128(
      (as1_vec4i*)(uintptr_t)g, (as3_vec4i*)(uint32_t)(uintptr_t)l, 0, 0);
#else
  *(int4*)l = *(const int4*)g;
#endif
}
__device__ __forceinline__ void async_wait() {
#if defined(__HIP_DEVICE_COMPILE__) && __has_builtin(__builtin_amdgcn_s_wait_asynccnt)
  __builtin_amdgcn_s_wait_asynccnt(0);
#elif defined(__AMDGCN__)
  asm volatile("s_wait_asynccnt 0" ::: "memory");
#endif
}

__global__ void prep_weights(const float* __restrict__ w1, const float* __restrict__ w2,
                             const float* __restrict__ w3, const float* __restrict__ w4,
                             const float* __restrict__ wfc1, const float* __restrict__ wfc2,
                             uint32_t* __restrict__ wmask1, int* __restrict__ wb2,
                             int* __restrict__ wb3, int* __restrict__ wb4,
                             int* __restrict__ wf1, int* __restrict__ wf2) {
  int blk = blockIdx.x;
  int tid = threadIdx.x;
  if (blk == 38) {
    if (tid < 32) {
      uint32_t m = 0;
      for (int j = 0; j < 27; ++j)
        if (w1[tid * 27 + j] > 0.f) m |= (1u << j);
      wmask1[tid] = m;
    }
    return;
  }
  const float* w;
  int* dst;
  int step, ntile, NT, Kact, Nact, kind;
  if (blk < 10)      { w = w2;   dst = wb2; step = blk / 2;        ntile = blk % 2;        NT = 2; Kact = 288; Nact = 32;  kind = 0; }
  else if (blk < 20) { w = w3;   dst = wb3; step = (blk - 10) / 2; ntile = (blk - 10) % 2; NT = 2; Kact = 288; Nact = 32;  kind = 0; }
  else if (blk < 28) { w = w4;   dst = wb4; step = (blk - 20) / 4; ntile = (blk - 20) % 4; NT = 4; Kact = 128; Nact = 64;  kind = 1; }
  else if (blk < 36) { w = wfc1; dst = wf1; step = 0;              ntile = blk - 28;       NT = 8; Kact = 64;  Nact = 128; kind = 2; }
  else               { w = wfc2; dst = wf2; step = blk - 36;       ntile = 0;              NT = 1; Kact = 128; Nact = 2;   kind = 3; }
  int lane = tid >> 3, v = tid & 7;
  int hi = (lane >= 16) ? 1 : 0;
  int n = ntile * 16 + (lane & 15);
  int kb = 64 * step + kbaseB(v, hi);
  uint32_t packed = 0;
  for (int b = 0; b < 4; ++b) {
    int k = kb + b;
    signed char val = 0;
    if (k < Kact && n < Nact) {
      int widx;
      if (kind == 0) {
        int c = k & 31;
        int sp = k >> 5;
        widx = n * 288 + c * 9 + sp;
      } else if (kind == 1) widx = n * 128 + k;
      else if (kind == 2)   widx = n * 64 + k;
      else                  widx = n * 128 + k;
      val = sgn8(w[widx]);
    }
    packed |= ((uint32_t)(uint8_t)val) << (8 * b);
  }
  dst[((step * NT + ntile) * 32 + lane) * 8 + v] = (int)packed;
}

__global__ void conv1_kernel(const float* __restrict__ x, const uint32_t* __restrict__ wmask,
                             const float* __restrict__ s1, const float* __restrict__ t1,
                             signed char* __restrict__ u2) {
  __shared__ uint8_t ss[720];
  __shared__ uint32_t pk[135];
  int img = blockIdx.x / 15;
  int py = blockIdx.x % 15;
  int tid = threadIdx.x;
  int r0 = 3 * py;
  for (int t = tid; t < 720; t += 512) {
    int c = t / 240, rem = t % 240, r = rem / 48, col = rem % 48;
    float v = x[((img * 3 + c) * 48 + (r0 + r)) * 48 + col];
    ss[t] = v > 0.f ? 1 : 0;
  }
  __syncthreads();
  for (int t = tid; t < 135; t += 512) {
    int cr = t / 45, cc = t % 45;
    uint32_t bits = 0;
#pragma unroll
    for (int c = 0; c < 3; ++c)
#pragma unroll
      for (int ky = 0; ky < 3; ++ky)
#pragma unroll
        for (int kx = 0; kx < 3; ++kx)
          if (ss[c * 240 + (cr + ky) * 48 + (cc + kx)]) bits |= 1u << (c * 9 + ky * 3 + kx);
    pk[t] = bits;
  }
  __syncthreads();
  if (tid < 480) {
    int px = tid >> 5, ch = tid & 31;
    uint32_t m = wmask[ch];
    int best = -1000;
#pragma unroll
    for (int dy = 0; dy < 3; ++dy)
#pragma unroll
      for (int dx = 0; dx < 3; ++dx) {
        int v = 27 - 2 * __popc(pk[dy * 45 + 3 * px + dx] ^ m);
        best = v > best ? v : best;
      }
    float f = s1[ch] * (float)best + t1[ch];
    u2[((img * 15 + py) * 15 + px) * 32 + ch] = f > 0.f ? 1 : -1;
  }
}

__global__ void conv2_kernel(const signed char* __restrict__ u2, const int* __restrict__ wb2,
                             const float* __restrict__ s2, const float* __restrict__ t2,
                             signed char* __restrict__ u3) {
  __shared__ int su2[1800];
  __shared__ int sconv[169 * 32 + 32];
  int img = blockIdx.x;
  int tid = threadIdx.x;
  {
    const char* gb = (const char*)(u2 + img * 7200);
    if (tid < 450) async_copy16(gb + tid * 16, (char*)su2 + tid * 16);
    async_wait();
  }
  __syncthreads();
  int lane = tid & 31, wave = tid >> 5;
  int hi = lane >> 4, lm = lane & 15;
  for (int t = wave; t < 22; t += 16) {
    int mtile = t >> 1, ntile = t & 1;
    int M = mtile * 16 + lm;
    int Mc = M < 169 ? M : 168;
    int py = Mc / 13, px = Mc % 13;
    v8i acc = {};
#pragma unroll
    for (int s = 0; s < 5; ++s) {
      v8i a;
#pragma unroll
      for (int v = 0; v < 8; ++v) {
        int k = 64 * s + kbaseA(v, hi);
        int val = 0;
        if (k < 288) {
          int sp = k >> 5, c = k & 31;
          int ky = sp / 3, kx = sp % 3;
          val = su2[((py + ky) * 15 + (px + kx)) * 8 + (c >> 2)];
        }
        a[v] = val;
      }
      const int* bp = wb2 + ((s * 2 + ntile) * 32 + lane) * 8;
      v8i b;
#pragma unroll
      for (int v = 0; v < 8; ++v) b[v] = bp[v];
      acc = __builtin_amdgcn_wmma_i32_16x16x64_iu8(true, a, true, b, acc, false, false);
    }
    int col = ntile * 16 + lm;
#pragma unroll
    for (int gR = 0; gR < 8; ++gR) {
      int row = mtile * 16 + gR + 8 * hi;
      int addr = (row < 169) ? (row * 32 + col) : (169 * 32 + lane);
      sconv[addr] = acc[gR];
    }
  }
  __syncthreads();
  if (tid < 512) {
    int qy = tid >> 7, qx = (tid >> 5) & 3, ch = tid & 31;
    int best = -100000;
#pragma unroll
    for (int dy = 0; dy < 3; ++dy)
#pragma unroll
      for (int dx = 0; dx < 3; ++dx) {
        int v = sconv[((3 * qy + dy) * 13 + (3 * qx + dx)) * 32 + ch];
        best = v > best ? v : best;
      }
    float f = s2[ch] * (float)best + t2[ch];
    u3[img * 512 + (qy * 4 + qx) * 32 + ch] = f > 0.f ? 1 : -1;
  }
}

__global__ void conv3_kernel(const signed char* __restrict__ u3, const int* __restrict__ wb3,
                             const float* __restrict__ s3, const float* __restrict__ t3,
                             signed char* __restrict__ u4) {
  __shared__ int su3[512];
  int img0 = blockIdx.x * 4;
  int tid = threadIdx.x;
  {
    const char* gb = (const char*)(u3 + img0 * 512);
    for (int t = tid; t < 128; t += 64) async_copy16(gb + t * 16, (char*)su3 + t * 16);
    async_wait();
  }
  __syncthreads();
  int lane = tid & 31, ntile = tid >> 5;
  int hi = lane >> 4, lm = lane & 15;
  int img_sub = lm >> 2, pos = lm & 3;
  int py = pos >> 1, px = pos & 1;
  v8i acc = {};
#pragma unroll
  for (int s = 0; s < 5; ++s) {
    v8i a;
#pragma unroll
    for (int v = 0; v < 8; ++v) {
      int k = 64 * s + kbaseA(v, hi);
      int val = 0;
      if (k < 288) {
        int sp = k >> 5, c = k & 31;
        int ky = sp / 3, kx = sp % 3;
        val = su3[img_sub * 128 + ((py + ky) * 4 + (px + kx)) * 8 + (c >> 2)];
      }
      a[v] = val;
    }
    const int* bp = wb3 + ((s * 2 + ntile) * 32 + lane) * 8;
    v8i b;
#pragma unroll
    for (int v = 0; v < 8; ++v) b[v] = bp[v];
    acc = __builtin_amdgcn_wmma_i32_16x16x64_iu8(true, a, true, b, acc, false, false);
  }
#pragma unroll
  for (int gR = 0; gR < 8; ++gR) {
    int M = gR + 8 * hi;
    int isub = M >> 2, p = M & 3;
    int ch = ntile * 16 + lm;
    float f = s3[ch] * (float)acc[gR] + t3[ch];
    u4[(img0 + isub) * 128 + ch * 4 + p] = f > 0.f ? 1 : -1;
  }
}

__global__ void conv4_kernel(const signed char* __restrict__ u4, const int* __restrict__ wb4,
                             const float* __restrict__ s4, const float* __restrict__ t4,
                             signed char* __restrict__ u5) {
  int mtile = blockIdx.x;
  int tid = threadIdx.x;
  int lane = tid & 31, ntile = tid >> 5;
  int hi = lane >> 4, lm = lane & 15;
  int row = mtile * 16 + lm;
  const int* arow = (const int*)(u4 + row * 128);
  v8i acc = {};
#pragma unroll
  for (int s = 0; s < 2; ++s) {
    v8i a;
#pragma unroll
    for (int v = 0; v < 8; ++v) a[v] = arow[(64 * s + kbaseA(v, hi)) >> 2];
    const int* bp = wb4 + ((s * 4 + ntile) * 32 + lane) * 8;
    v8i b;
#pragma unroll
    for (int v = 0; v < 8; ++v) b[v] = bp[v];
    acc = __builtin_amdgcn_wmma_i32_16x16x64_iu8(true, a, true, b, acc, false, false);
  }
#pragma unroll
  for (int gR = 0; gR < 8; ++gR) {
    int r = mtile * 16 + gR + 8 * hi;
    int o = ntile * 16 + lm;
    float f = s4[o] * (float)acc[gR] + t4[o];
    u5[r * 64 + o] = (f - 0.01f) > 0.f ? 1 : -1;
  }
}

__global__ void fc1_kernel(const signed char* __restrict__ u5, const int* __restrict__ wf1,
                           const float* __restrict__ s5, const float* __restrict__ t5,
                           signed char* __restrict__ u6) {
  int mtile = blockIdx.x;
  int tid = threadIdx.x;
  int lane = tid & 31, ntile = tid >> 5;
  int hi = lane >> 4, lm = lane & 15;
  int row = mtile * 16 + lm;
  const int* arow = (const int*)(u5 + row * 64);
  v8i a;
#pragma unroll
  for (int v = 0; v < 8; ++v) a[v] = arow[kbaseA(v, hi) >> 2];
  const int* bp = wf1 + (ntile * 32 + lane) * 8;
  v8i b;
#pragma unroll
  for (int v = 0; v < 8; ++v) b[v] = bp[v];
  v8i acc = {};
  acc = __builtin_amdgcn_wmma_i32_16x16x64_iu8(true, a, true, b, acc, false, false);
#pragma unroll
  for (int gR = 0; gR < 8; ++gR) {
    int r = mtile * 16 + gR + 8 * hi;
    int n = ntile * 16 + lm;
    float f = s5[n] * (float)acc[gR] + t5[n];
    u6[r * 128 + n] = (f - 0.01f) > 0.f ? 1 : -1;
  }
}

__global__ void fc2_kernel(const signed char* __restrict__ u6, const int* __restrict__ wf2,
                           float* __restrict__ out) {
  int mtile = blockIdx.x;
  int lane = threadIdx.x & 31;
  int hi = lane >> 4, lm = lane & 15;
  int row = mtile * 16 + lm;
  const int* arow = (const int*)(u6 + row * 128);
  v8i acc = {};
#pragma unroll
  for (int s = 0; s < 2; ++s) {
    v8i a;
#pragma unroll
    for (int v = 0; v < 8; ++v) a[v] = arow[(64 * s + kbaseA(v, hi)) >> 2];
    const int* bp = wf2 + (s * 32 + lane) * 8;
    v8i b;
#pragma unroll
    for (int v = 0; v < 8; ++v) b[v] = bp[v];
    acc = __builtin_amdgcn_wmma_i32_16x16x64_iu8(true, a, true, b, acc, false, false);
  }
  if (lm < 2) {
#pragma unroll
    for (int gR = 0; gR < 8; ++gR) {
      int r = mtile * 16 + gR + 8 * hi;
      out[r * 2 + lm] = (float)acc[gR];
    }
  }
}

extern "C" void kernel_launch(void* const* d_in, const int* in_sizes, int n_in,
                              void* d_out, int out_size, void* d_ws, size_t ws_size,
                              hipStream_t stream) {
  const float* x    = (const float*)d_in[0];
  const float* w1   = (const float*)d_in[1];
  const float* w2   = (const float*)d_in[2];
  const float* w3   = (const float*)d_in[3];
  const float* w4   = (const float*)d_in[4];
  const float* wfc1 = (const float*)d_in[5];
  const float* wfc2 = (const float*)d_in[6];
  const float* s1 = (const float*)d_in[7];
  const float* t1 = (const float*)d_in[8];
  const float* s2 = (const float*)d_in[9];
  const float* t2 = (const float*)d_in[10];
  const float* s3 = (const float*)d_in[11];
  const float* t3 = (const float*)d_in[12];
  const float* s4 = (const float*)d_in[13];
  const float* t4 = (const float*)d_in[14];
  const float* s5 = (const float*)d_in[15];
  const float* t5 = (const float*)d_in[16];
  float* out = (float*)d_out;

  char* ws = (char*)d_ws;
  signed char* u2 = (signed char*)(ws + 0);
  signed char* u3 = (signed char*)(ws + 29491200);
  signed char* u4 = (signed char*)(ws + 31588352);
  signed char* u5 = (signed char*)(ws + 32112640);
  signed char* u6 = (signed char*)(ws + 32374784);
  uint32_t* wmask1 = (uint32_t*)(ws + 32899072);
  int* wb2 = (int*)(ws + 32899328);
  int* wb3 = (int*)(ws + 32909568);
  int* wb4 = (int*)(ws + 32919808);
  int* wf1 = (int*)(ws + 32928000);
  int* wf2 = (int*)(ws + 32936192);

  prep_weights<<<39, 256, 0, stream>>>(w1, w2, w3, w4, wfc1, wfc2,
                                       wmask1, wb2, wb3, wb4, wf1, wf2);
  conv1_kernel<<<4096 * 15, 512, 0, stream>>>(x, wmask1, s1, t1, u2);
  conv2_kernel<<<4096, 512, 0, stream>>>(u2, wb2, s2, t2, u3);
  conv3_kernel<<<1024, 64, 0, stream>>>(u3, wb3, s3, t3, u4);
  conv4_kernel<<<256, 128, 0, stream>>>(u4, wb4, s4, t4, u5);
  fc1_kernel<<<256, 256, 0, stream>>>(u5, wf1, s5, t5, u6);
  fc2_kernel<<<256, 32, 0, stream>>>(u6, wf2, out);
}
